// Lstm_Trio_Attn_BERT_12395275616753
// MI455X (gfx1250) — matrix operand order repaired and checked by reading, not compiled, not run on hardware
//
#include <hip/hip_runtime.h>
#include <hip/hip_bf16.h>
#include <math.h>

// Problem constants
#define B_  64
#define S_  256
#define H_  768
#define BS_ (B_*S_)
#define N2_ 1536   // concatenated [W_r|W_t] columns / [W_p;W_x] rows

typedef __bf16 bf16_t;
typedef __attribute__((ext_vector_type(4)))  bf16_t v4bf;
typedef __attribute__((ext_vector_type(8)))  bf16_t v8bf;
typedef __attribute__((ext_vector_type(16))) bf16_t v16bf;
typedef __attribute__((ext_vector_type(8)))  float  v8f;
typedef __attribute__((ext_vector_type(4)))  float  v4f;

__device__ __forceinline__ v8f zero_v8f() {
  v8f z;
#pragma unroll
  for (int i = 0; i < 8; ++i) z[i] = 0.0f;
  return z;
}

__device__ __forceinline__ v16bf cat8(v8bf lo, v8bf hi) {
  return __builtin_shufflevector(lo, hi, 0,1,2,3,4,5,6,7,8,9,10,11,12,13,14,15);
}

// Hardware tanh if the toolchain exposes it (gfx1250 has V_TANH_F32 as a
// TRANS op); otherwise fall back to libm.
__device__ __forceinline__ float fast_tanh(float x) {
#if defined(__AMDGCN__) && __has_builtin(__builtin_amdgcn_tanhf)
  return __builtin_amdgcn_tanhf(x);
#elif defined(__AMDGCN__) && __has_builtin(__builtin_amdgcn_tanh_f32)
  return __builtin_amdgcn_tanh_f32(x);
#else
  return tanhf(x);
#endif
}

// ---------------------------------------------------------------------------
// K0: pre-transpose + f32->bf16 weight conversion.
//   Wrt[n][k] = (n<768 ? W_r[k][n] : W_t[k][n-768])   (n in [0,1536), k in [0,768))
//   Wpx[n][k] = (k<768 ? W_p[k][n] : W_x[k-768][n])   (n in [0,768),  k in [0,1536))
// Transposed [n][k] layout gives per-lane K-contiguous 32B B-fragment reads.
// ---------------------------------------------------------------------------
__global__ void convert_weights(const float* __restrict__ Wr, const float* __restrict__ Wt,
                                const float* __restrict__ Wp, const float* __restrict__ Wx,
                                bf16_t* __restrict__ Wrt, bf16_t* __restrict__ Wpx) {
  int idx = blockIdx.x * blockDim.x + threadIdx.x;
  if (idx >= N2_ * H_) return;
  int n = idx / H_, k = idx % H_;
  float v = (n < H_) ? Wr[k * H_ + n] : Wt[k * H_ + (n - H_)];
  Wrt[idx] = (bf16_t)v;
  int n2 = idx / N2_, k2 = idx % N2_;
  float v2 = (k2 < H_) ? Wp[k2 * H_ + n2] : Wx[(k2 - H_) * H_ + n2];
  Wpx[idx] = (bf16_t)v2;
}

// ---------------------------------------------------------------------------
// K1: 128x128-tile bf16 WMMA GEMM, M=16384 N=768 K=768.
// 8 waves (2x4); each wave owns a 64x32 sub-tile = 4x2 wmma accumulators.
// MODE 0: Wy = Y @ W_y, stored bf16 row-major [b*S+s][n]
// MODE 1: Hh = hs @ W_h, stored f32 remapped  [s*B+b][n]
// MODE is a template parameter so the epilogue is branch-free.
// ---------------------------------------------------------------------------
template <int MODE>
__global__ __launch_bounds__(256) void gemm128_wmma(
    const float* __restrict__ Ain, const float* __restrict__ Wmat,
    bf16_t* __restrict__ outBF, float* __restrict__ outRemapF) {
  __shared__ __align__(16) bf16_t At[128 * 40];  // [row][k], padded 32->40
  __shared__ __align__(16) bf16_t Bt[128 * 40];  // [n][k] transposed, padded

  const int tid = threadIdx.x, lane = tid & 31, wv = tid >> 5;
  const int nblk  = H_ / 128;                 // 6
  const int mtile = blockIdx.x / nblk;
  const int ntile = blockIdx.x % nblk;
  const int m0 = mtile * 128, n0 = ntile * 128;
  const int m0w = (wv >> 2) * 64, n0w = (wv & 3) * 32;

  v8f c[4][2];
#pragma unroll
  for (int i = 0; i < 4; ++i)
#pragma unroll
    for (int j = 0; j < 2; ++j) c[i][j] = zero_v8f();

  for (int k0 = 0; k0 < H_; k0 += 32) {
    __syncthreads();
    // Prefetch next K-tile into cache while we stage/compute this one.
    if (k0 + 32 < H_) {
      __builtin_prefetch(Ain + (size_t)(m0 + (tid >> 1)) * H_ + k0 + 32, 0, 1);
      __builtin_prefetch(Wmat + (size_t)(k0 + 32 + (tid >> 5)) * H_ + n0 + (tid & 31) * 4, 0, 1);
    }
    // Stage A tile (128x32 f32 -> bf16): coalesced float4 loads, packed 8B DS stores
#pragma unroll
    for (int p = 0; p < 4; ++p) {
      int row = p * 32 + (tid >> 3);
      int col = (tid & 7) * 4;
      v4f a4 = *(const v4f*)(Ain + (size_t)(m0 + row) * H_ + k0 + col);
      v4bf a4b;
#pragma unroll
      for (int e = 0; e < 4; ++e) a4b[e] = (bf16_t)a4[e];
      *(v4bf*)(At + row * 40 + col) = a4b;     // 8B aligned: row*80 + col*2
    }
    // Stage B tile transposed (32x128 -> Bt[n][k]); scalar stores (transpose)
#pragma unroll
    for (int p = 0; p < 4; ++p) {
      int k  = p * 8 + (tid >> 5);
      int n4 = (tid & 31) * 4;
      v4f b4 = *(const v4f*)(Wmat + (size_t)(k0 + k) * H_ + n0 + n4);
#pragma unroll
      for (int e = 0; e < 4; ++e) Bt[(n4 + e) * 40 + k] = (bf16_t)b4[e];
    }
    __syncthreads();

    const int ka = 8 * (lane >> 4);    // A: K chunks at {ka, ka+16}
    const int kb = 8 * (lane >> 4);    // B: K chunks at {kb, kb+16}
    v16bf afr[4];
#pragma unroll
    for (int i = 0; i < 4; ++i) {
      int row = m0w + i * 16 + (lane & 15);
      v8bf lo = *(const v8bf*)(At + row * 40 + ka);
      v8bf hi = *(const v8bf*)(At + row * 40 + ka + 16);
      afr[i] = cat8(lo, hi);
    }
#pragma unroll
    for (int j = 0; j < 2; ++j) {
      int row = n0w + j * 16 + (lane & 15);
      v8bf lo = *(const v8bf*)(Bt + row * 40 + kb);
      v8bf hi = *(const v8bf*)(Bt + row * 40 + kb + 16);
      v16bf bfr = cat8(lo, hi);
#pragma unroll
      for (int i = 0; i < 4; ++i)
        c[i][j] = __builtin_amdgcn_wmma_f32_16x16x32_bf16(
            false, afr[i], false, bfr, (short)0, c[i][j], false, false);
    }
  }

  // Epilogue: C/D layout M = e + 8*(lane>=16), N = lane%16  (branch-free)
#pragma unroll
  for (int i = 0; i < 4; ++i)
#pragma unroll
    for (int j = 0; j < 2; ++j) {
      int ng = n0 + n0w + j * 16 + (lane & 15);
#pragma unroll
      for (int e = 0; e < 8; ++e) {
        int mg = m0 + m0w + i * 16 + e + 8 * (lane >> 4);
        if (MODE == 0) {
          outBF[(size_t)mg * H_ + ng] = (bf16_t)c[i][j][e];
        } else {
          int bb = mg >> 8, ss = mg & 255;              // mg = b*S + s
          outRemapF[(size_t)(ss * B_ + bb) * H_ + ng] = c[i][j][e];
        }
      }
    }
}

// ---------------------------------------------------------------------------
// K2: sequential attention scan; one workgroup (8 waves) per batch element.
// Phase A: [r@W_r | r@W_t] via broadcast-A WMMA (all 16 rows = r slice).
// Phase B: scores_s = sum_h tanh(Wy+Wh)*w  (bf16 Wy from L2, shfl reduce).
// Phase C: softmax over S=256 (1 thread per s, wave32 shfl + LDS partials).
// Phase D: r_t = alpha.Y + tanh(r@W_t); masked update of r.
// ---------------------------------------------------------------------------
__global__ __launch_bounds__(256) void recurrence_kernel(
    const bf16_t* __restrict__ Wy, const float* __restrict__ Hh,
    const float* __restrict__ Y, const float* __restrict__ maskY,
    const float* __restrict__ maskH, const float* __restrict__ wvecg,
    const bf16_t* __restrict__ Wrt, const float* __restrict__ hs,
    bf16_t* __restrict__ A2) {
  __shared__ float r_f[H_];
  __shared__ __align__(16) bf16_t r_b[H_];
  __shared__ float Wh[H_];
  __shared__ float rt_t[H_];
  __shared__ float wv_l[H_];
  __shared__ float sc[S_];
  __shared__ float neg[S_];
  __shared__ float red[8];

  const int b = blockIdx.x;
  const int tid = threadIdx.x, lane = tid & 31, wave = tid >> 5;

  for (int h = tid; h < H_; h += 256) { r_f[h] = 0.f; r_b[h] = (bf16_t)0.f; wv_l[h] = wvecg[h]; }
  for (int s = tid; s < S_; s += 256) neg[s] = -1000.0f * (1.0f - maskY[b * S_ + s]);
  __syncthreads();

  for (int t = 0; t < S_; ++t) {
    // ---- Phase A: broadcast-A WMMA vec-mat, N2=1536 -> 96 n-subtiles / 8 waves
    {
      const int ka = 8 * (lane >> 4);
      const int kb = 8 * (lane >> 4);
      for (int nt = wave; nt < N2_ / 16; nt += 8) {
        v8f acc = zero_v8f();
        const bf16_t* brow = Wrt + (size_t)(nt * 16 + (lane & 15)) * H_;
        for (int k0 = 0; k0 < H_; k0 += 32) {
          v8bf alo = *(const v8bf*)(r_b + k0 + ka);
          v8bf ahi = *(const v8bf*)(r_b + k0 + ka + 16);
          v8bf blo = *(const v8bf*)(brow + k0 + kb);
          v8bf bhi = *(const v8bf*)(brow + k0 + kb + 16);
          acc = __builtin_amdgcn_wmma_f32_16x16x32_bf16(
              false, cat8(alo, ahi), false, cat8(blo, bhi), (short)0, acc, false, false);
        }
        float val = acc[0];             // all 16 output rows identical (broadcast A)
        int n = nt * 16 + (lane & 15);
        if (lane < 16) {
          if (n < H_) Wh[n] = Hh[((size_t)t * B_ + b) * H_ + n] + val;
          else        rt_t[n - H_] = fast_tanh(val);
        }
      }
    }
    __syncthreads();

    // ---- Phase B: attention scores, lanes cooperate over H
    for (int s = wave; s < S_; s += 8) {
      const bf16_t* wyrow = Wy + ((size_t)b * S_ + s) * H_;
      float acc = 0.f;
#pragma unroll
      for (int j = 0; j < H_; j += 256) {
        int h = j + lane * 8;
        v8bf wy8 = *(const v8bf*)(wyrow + h);   // 16B coalesced, L2-resident
#pragma unroll
        for (int e = 0; e < 8; ++e)
          acc += fast_tanh((float)wy8[e] + Wh[h + e]) * wv_l[h + e];
      }
#pragma unroll
      for (int off = 16; off; off >>= 1) acc += __shfl_xor(acc, off, 32);
      if (lane == 0) sc[s] = acc + neg[s];
    }
    __syncthreads();

    // ---- Phase C: softmax over S (1 thread per score)
    {
      float v = sc[tid];
      float mx = v;
#pragma unroll
      for (int off = 16; off; off >>= 1) mx = fmaxf(mx, __shfl_xor(mx, off, 32));
      if (lane == 0) red[wave] = mx;
      __syncthreads();
      float M = red[0];
#pragma unroll
      for (int i = 1; i < 8; ++i) M = fmaxf(M, red[i]);
      __syncthreads();
      float e = __expf(v - M);
      float sm = e;
#pragma unroll
      for (int off = 16; off; off >>= 1) sm += __shfl_xor(sm, off, 32);
      if (lane == 0) red[wave] = sm;
      __syncthreads();
      float tot = 0.f;
#pragma unroll
      for (int i = 0; i < 8; ++i) tot += red[i];
      sc[tid] = e / tot;                 // alpha
    }
    __syncthreads();

    // ---- Phase D: r_t = alpha.Y + tanh(r@W_t); masked update
    {
      float a0 = 0.f, a1 = 0.f, a2 = 0.f;
      const float* Yb = Y + (size_t)b * S_ * H_;
      for (int s = 0; s < S_; ++s) {
        float al = sc[s];
        const float* yr = Yb + (size_t)s * H_;
        a0 = fmaf(al, yr[tid],       a0);     // coalesced across threads
        a1 = fmaf(al, yr[tid + 256], a1);
        a2 = fmaf(al, yr[tid + 512], a2);
      }
      float m = maskH[b * S_ + t];
#pragma unroll
      for (int q = 0; q < 3; ++q) {
        int h = tid + q * 256;
        float rt = (q == 0 ? a0 : (q == 1 ? a1 : a2)) + rt_t[h];
        float rn = rt * m + r_f[h] * (1.0f - m);
        r_f[h] = rn;
        r_b[h] = (bf16_t)rn;
      }
    }
    __syncthreads();
  }

  // A2 = [r_final | hs_last] (bf16, 64x1536) for the fused final GEMM
  for (int k = tid; k < N2_; k += 256) {
    float v = (k < H_) ? r_f[k] : hs[((size_t)b * S_ + (S_ - 1)) * H_ + (k - H_)];
    A2[(size_t)b * N2_ + k] = (bf16_t)v;
  }
}

// ---------------------------------------------------------------------------
// K3: h_star = tanh(A2 @ [W_p;W_x]) : 64x768x1536 WMMA GEMM, 1 tile/wave.
// ---------------------------------------------------------------------------
__global__ __launch_bounds__(256) void final_gemm(
    const bf16_t* __restrict__ A2, const bf16_t* __restrict__ Wpx,
    float* __restrict__ out) {
  const int tid = threadIdx.x, lane = tid & 31, wave = tid >> 5;
  const int g = blockIdx.x * 8 + wave;      // 0..191 = 4 mtiles x 48 ntiles
  const int mtile = g / 48, ntile = g % 48;
  const int m0 = mtile * 16, n0 = ntile * 16;
  const int ka = 8 * (lane >> 4), kb = 8 * (lane >> 4);
  const bf16_t* arow = A2  + (size_t)(m0 + (lane & 15)) * N2_;
  const bf16_t* brow = Wpx + (size_t)(n0 + (lane & 15)) * N2_;
  v8f acc = zero_v8f();
  for (int k0 = 0; k0 < N2_; k0 += 32) {
    v8bf alo = *(const v8bf*)(arow + k0 + ka);
    v8bf ahi = *(const v8bf*)(arow + k0 + ka + 16);
    v8bf blo = *(const v8bf*)(brow + k0 + kb);
    v8bf bhi = *(const v8bf*)(brow + k0 + kb + 16);
    acc = __builtin_amdgcn_wmma_f32_16x16x32_bf16(
        false, cat8(alo, ahi), false, cat8(blo, bhi), (short)0, acc, false, false);
  }
  int n = n0 + (lane & 15);
#pragma unroll
  for (int e = 0; e < 8; ++e) {
    int m = m0 + e + 8 * (lane >> 4);
    out[(size_t)m * H_ + n] = fast_tanh(acc[e]);
  }
}

// ---------------------------------------------------------------------------
extern "C" void kernel_launch(void* const* d_in, const int* in_sizes, int n_in,
                              void* d_out, int out_size, void* d_ws, size_t ws_size,
                              hipStream_t stream) {
  (void)in_sizes; (void)n_in; (void)out_size; (void)ws_size;
  const float* Y   = (const float*)d_in[0];
  const float* mY  = (const float*)d_in[1];
  const float* HS  = (const float*)d_in[2];
  const float* mH  = (const float*)d_in[3];
  const float* Wym = (const float*)d_in[4];
  const float* Whm = (const float*)d_in[5];
  const float* Wr  = (const float*)d_in[6];
  const float* wv  = (const float*)d_in[7];
  const float* Wt  = (const float*)d_in[8];
  const float* Wp  = (const float*)d_in[9];
  const float* Wx  = (const float*)d_in[10];

  // Workspace layout (bytes, 16B-aligned offsets), total ~80.5 MB
  char* ws = (char*)d_ws;
  bf16_t* WyP  = (bf16_t*)(ws);               // Wy  bf16 [16384][768]   25,165,824
  float*  HhP  = (float*)(ws + 25165824);     // Hh  f32  [256][64][768] 50,331,648
  bf16_t* WrtP = (bf16_t*)(ws + 75497472);    // [Wr|Wt]^T bf16 [1536][768]  2,359,296
  bf16_t* WpxP = (bf16_t*)(ws + 77856768);    // [Wp;Wx]^T bf16 [768][1536]  2,359,296
  bf16_t* A2P  = (bf16_t*)(ws + 80216064);    // [r|h_last] bf16 [64][1536]    196,608

  convert_weights<<<(N2_ * H_ + 255) / 256, 256, 0, stream>>>(Wr, Wt, Wp, Wx, WrtP, WpxP);
  gemm128_wmma<0><<<(BS_ / 128) * (H_ / 128), 256, 0, stream>>>(Y,  Wym, WyP, nullptr);
  gemm128_wmma<1><<<(BS_ / 128) * (H_ / 128), 256, 0, stream>>>(HS, Whm, nullptr, HhP);
  recurrence_kernel<<<B_, 256, 0, stream>>>(WyP, HhP, Y, mY, mH, wv, WrtP, HS, A2P);
  final_gemm<<<24, 256, 0, stream>>>(A2P, WpxP, (float*)d_out);
}
